// circleLOIMLossSafe_79654463471716
// MI455X (gfx1250) — compile-verified, matrix operand order checked by reading, not run on hardware
//
#include <hip/hip_runtime.h>
#include <hip/hip_bf16.h>

// ---------------- problem constants ----------------
#define NN 1024
#define FF 256
#define PP 32768
#define QQ 8192
#define FFP 264   // padded LDS row stride (halfs): 132 dwords == 4 mod 64 banks

typedef __bf16 v16bf __attribute__((ext_vector_type(16)));
typedef float  v8f   __attribute__((ext_vector_type(8)));

union Frag { v16bf v; uint4 q[2]; };

__device__ __forceinline__ unsigned short f2bf(float f) {
    unsigned int u = __float_as_uint(f);
    unsigned int r = (u + 0x7FFFu + ((u >> 16) & 1u)) >> 16;   // RNE
    return (unsigned short)r;
}

// ---------------- setup: zero rowsum/loss, compute mean(ious)<0.2 flag ----------------
__global__ void k_setup(const float* __restrict__ ious, float* __restrict__ rowsum,
                        float* __restrict__ loss, int* __restrict__ flag) {
    __shared__ float sh[256];
    int t = threadIdx.x;
    float s = 0.f;
    for (int i = t; i < NN; i += 256) { rowsum[i] = 0.f; s += ious[i]; }
    sh[t] = s; __syncthreads();
    for (int k = 128; k > 0; k >>= 1) { if (t < k) sh[t] += sh[t + k]; __syncthreads(); }
    if (t == 0) { loss[0] = 0.f; flag[0] = (sh[0] * (1.f / NN) < 0.2f) ? 1 : 0; }
}

// ---------------- normalize x rows; emit fp32 + bf16 copies ----------------
__global__ __launch_bounds__(256) void k_normalize(const float* __restrict__ in,
                                                   float* __restrict__ xs,
                                                   unsigned short* __restrict__ xh) {
    __shared__ float sh[256];
    int r = blockIdx.x, t = threadIdx.x;
    float v = in[r * FF + t];
    sh[t] = v * v; __syncthreads();
    for (int k = 128; k > 0; k >>= 1) { if (t < k) sh[t] += sh[t + k]; __syncthreads(); }
    float nrm = sqrtf(sh[0]);
    nrm = nrm < 1e-12f ? 1e-12f : nrm;
    float o = v / nrm;
    xs[r * FF + t] = o;
    xh[r * FF + t] = f2bf(o);
}

// ---------------- convert weight matrix to bf16 + all-zero-row flags (4 rows/block) ----------------
__global__ __launch_bounds__(256) void k_cvt(const float* __restrict__ src,
                                             unsigned short* __restrict__ dst,
                                             unsigned char* __restrict__ bad) {
    __shared__ int nz4[4];
    int t = threadIdx.x;
    int sub = t >> 6;                   // row within this block's group of 4
    int e4  = (t & 63) * 4;             // element offset within row
    size_t row = (size_t)blockIdx.x * 4 + sub;
    if (t < 4) nz4[t] = 0;
    __syncthreads();
    float4 v = *(const float4*)(src + row * FF + e4);
    if (v.x != 0.f || v.y != 0.f || v.z != 0.f || v.w != 0.f) nz4[sub] = 1;
    ushort4 o;
    o.x = f2bf(v.x); o.y = f2bf(v.y); o.z = f2bf(v.z); o.w = f2bf(v.w);
    *(ushort4*)(dst + row * FF + e4) = o;
    __syncthreads();
    if (t < 4) bad[blockIdx.x * 4 + t] = (nz4[t] == 0) ? 1 : 0;
}

// ---------------- WMMA GEMM + score transform + streaming sum(exp(proj-30)) ----------------
// grid: (colGroups, 16 m-blocks of 64 rows), 256 threads = 8 waves.
// Per column tile: load all 8 B fragments as one clause (latencies overlap, one
// amortized L2 wait), then 32 WMMAs reusing each B fragment across 4 M-tiles.
template <int MODE>
__global__ __launch_bounds__(256) void k_gemm(const unsigned short* __restrict__ xh,
                                              const unsigned short* __restrict__ wh,
                                              const unsigned char* __restrict__ bad,
                                              float* __restrict__ rowsum) {
    __shared__ __align__(16) unsigned short As[64 * FFP];

    const int tid   = threadIdx.x;
    const int lane  = tid & 31;
    const int wv    = tid >> 5;
    const int n     = lane & 15;
    const int h     = lane >> 4;
    const int mrow0 = blockIdx.y * 64;

    // stage A (64x256 bf16) into LDS with padded row stride
    {
        const uint4* src = (const uint4*)(xh + (size_t)mrow0 * FF);
#pragma unroll
        for (int i = 0; i < 8; ++i) {
            int c = tid + i * 256;                 // 2048 16B chunks
            int r = c >> 5, off = (c & 31) * 8;
            *(uint4*)(As + r * FFP + off) = src[c];
        }
    }
    __syncthreads();

    float eacc[4][8] = {};

    for (int j = 0; j < 8; ++j) {
        const int tile = blockIdx.x * 64 + wv * 8 + j;
        const int col  = tile * 16 + n;            // B column = weight row
        const unsigned short* brow = wh + (size_t)col * FF;

        Frag b[8];
#pragma unroll
        for (int ks = 0; ks < 8; ++ks) {           // one clause: 16 b128 loads in flight
            b[ks].q[0] = *(const uint4*)(brow + ks * 32 + h * 8);
            b[ks].q[1] = *(const uint4*)(brow + ks * 32 + 16 + h * 8);
        }

        v8f c[4] = {};
#pragma unroll
        for (int ks = 0; ks < 8; ++ks) {
#pragma unroll
            for (int mt = 0; mt < 4; ++mt) {
                Frag a;
                const unsigned short* ar = As + (mt * 16 + n) * FFP;
                a.q[0] = *(const uint4*)(ar + ks * 32 + h * 8);
                a.q[1] = *(const uint4*)(ar + ks * 32 + 16 + h * 8);
                c[mt] = __builtin_amdgcn_wmma_f32_16x16x32_bf16(false, a.v, false, b[ks].v,
                                                                (short)0, c[mt], false, false);
            }
        }

        const bool isbad = bad[col] != 0;
#pragma unroll
        for (int mt = 0; mt < 4; ++mt) {
#pragma unroll
            for (int v = 0; v < 8; ++v) {
                float s = c[mt][v];
                float proj;
                if (MODE == 0) {                   // lut scores (bad-pos fixed in finalize)
                    if (isbad) proj = -30.f;
                    else {
                        float an = s + 0.1f; an = (an <= 0.f) ? 1e-6f : an;
                        proj = (s - 0.1f) * an * 30.f;
                    }
                } else {                           // cq scores (torch chained-assign bug)
                    proj = isbad ? -30.f : ((s <= 0.f ? 1e-6f : s) * 1e-6f) * 30.f;
                }
                eacc[mt][v] += __expf(proj - 30.f);  // fixed-shift LSE (proj <= 30)
            }
        }
    }
#pragma unroll
    for (int mt = 0; mt < 4; ++mt)
#pragma unroll
        for (int v = 0; v < 8; ++v) {
            float e = eacc[mt][v];
            e += __shfl_xor(e, 1, 32);
            e += __shfl_xor(e, 2, 32);
            e += __shfl_xor(e, 4, 32);
            e += __shfl_xor(e, 8, 32);             // reduce over n within 16-lane half
            if (n == 0) atomicAdd(&rowsum[mrow0 + mt * 16 + v + 8 * h], e);
        }
}

// ---------------- per-row NLL: logZ - proj[label]; bad-pos row-sum correction ----------------
__global__ __launch_bounds__(256) void k_finalize(const float* __restrict__ xs,
                                                  const float* __restrict__ lut,
                                                  const int* __restrict__ label,
                                                  const unsigned char* __restrict__ badl,
                                                  const float* __restrict__ rowsum,
                                                  float* __restrict__ loss) {
    __shared__ float sh[256];
    int r = blockIdx.x, t = threadIdx.x;
    int y = label[r];
    if (y >= PP) return;                            // ignore_index -> zero contribution
    sh[t] = xs[r * FF + t] * lut[(size_t)y * FF + t];
    __syncthreads();
    for (int k = 128; k > 0; k >>= 1) { if (t < k) sh[t] += sh[t + k]; __syncthreads(); }
    if (t == 0) {
        float v = sh[0], pl;
        float Z = rowsum[r];
        if (badl[y]) {
            float ap = 1.1f - v; ap = (ap <= 0.f) ? 1e-6f : ap;
            pl = -ap * 30.f;
            // GEMM counted this entry as exp(-60); swap in the true pos term
            Z += __expf(pl - 30.f) - 8.7565107627e-27f;
        } else {
            float an = v + 0.1f; an = (an <= 0.f) ? 1e-6f : an;
            pl = (v - 0.1f) * an * 30.f;
        }
        float logZ = 30.f + __logf(Z);
        atomicAdd(loss, (logZ - pl) * (1.f / NN));
    }
}

// ---------------- sequential lut update, parallel over label residues ----------------
__global__ __launch_bounds__(256) void k_update_lut(const float* __restrict__ xs,
                                                    const int* __restrict__ label,
                                                    const float* __restrict__ ious,
                                                    const int* __restrict__ flag,
                                                    float* __restrict__ lut) {
    if (!flag[0]) return;
    __shared__ float sh[256];
    const int own = blockIdx.x, t = threadIdx.x;
    for (int pass = 0; pass < 2; ++pass) {
        for (int i = 0; i < NN; ++i) {
            int y = label[i];
            if (y >= PP || (y & 255) != own) continue;   // uniform across block
            float m  = pass ? ious[i] : 0.5f;            // pass0: momentum, pass1: iou mix
            float rv = (1.f - m) * lut[(size_t)y * FF + t] + m * xs[i * FF + t];
            sh[t] = rv * rv; __syncthreads();
            for (int k = 128; k > 0; k >>= 1) { if (t < k) sh[t] += sh[t + k]; __syncthreads(); }
            float nrm = sqrtf(sh[0]);
            nrm = nrm < 1e-12f ? 1e-12f : nrm;
            __syncthreads();
            lut[(size_t)y * FF + t] = rv / nrm;
        }
    }
}

// ---------------- sequential circular-queue update (global sample order) ----------------
__global__ void k_update_cq(const float* __restrict__ xs, const int* __restrict__ label,
                            const int* __restrict__ flag, float* __restrict__ cq) {
    if (!flag[0]) return;
    int t = threadIdx.x, hdr = 0;
    for (int pass = 0; pass < 2; ++pass)
        for (int i = 0; i < NN; ++i) {
            int y = label[i];
            if (y < PP) continue;
            cq[(size_t)hdr * FF + t] = xs[i * FF + t];
            hdr = (hdr + 1) & (QQ - 1);
        }
}

extern "C" void kernel_launch(void* const* d_in, const int* in_sizes, int n_in,
                              void* d_out, int out_size, void* d_ws, size_t ws_size,
                              hipStream_t stream) {
    const float* inputs = (const float*)d_in[0];
    const int*   label  = (const int*)d_in[1];
    const float* ious   = (const float*)d_in[2];
    const float* lut    = (const float*)d_in[3];
    const float* cq     = (const float*)d_in[4];

    float* out_loss = (float*)d_out;
    float* out_lut  = out_loss + 1;
    float* out_cq   = out_lut + (size_t)PP * FF;

    char* ws = (char*)d_ws;
    float*          xs     = (float*)(ws);                       // 1,048,576 B
    unsigned short* xh     = (unsigned short*)(ws + 1048576);    //   524,288 B
    unsigned short* lh     = (unsigned short*)(ws + 1572864);    // 16,777,216 B
    unsigned short* ch     = (unsigned short*)(ws + 18350080);   //  4,194,304 B
    unsigned char*  badl   = (unsigned char*)(ws + 22544384);    //    32,768 B
    unsigned char*  badc   = (unsigned char*)(ws + 22577152);    //     8,192 B
    float*          rowsum = (float*)(ws + 22585344);            //     4,096 B
    int*            flag   = (int*)(ws + 22589440);

    // outputs start as copies of lut/cq (loss path reads the originals)
    hipMemcpyAsync(out_lut, lut, (size_t)PP * FF * sizeof(float), hipMemcpyDeviceToDevice, stream);
    hipMemcpyAsync(out_cq,  cq,  (size_t)QQ * FF * sizeof(float), hipMemcpyDeviceToDevice, stream);

    k_setup<<<1, 256, 0, stream>>>(ious, rowsum, out_loss, flag);
    k_normalize<<<NN, 256, 0, stream>>>(inputs, xs, xh);
    k_cvt<<<PP / 4, 256, 0, stream>>>(lut, lh, badl);
    k_cvt<<<QQ / 4, 256, 0, stream>>>(cq, ch, badc);

    k_gemm<0><<<dim3(PP / 16 / 64, NN / 64), 256, 0, stream>>>(xh, lh, badl, rowsum);
    k_gemm<1><<<dim3(QQ / 16 / 64, NN / 64), 256, 0, stream>>>(xh, ch, badc, rowsum);

    k_finalize<<<NN, 256, 0, stream>>>(xs, lut, label, badl, rowsum, out_loss);
    k_update_lut<<<256, 256, 0, stream>>>(xs, label, ious, flag, out_lut);
    k_update_cq<<<1, 256, 0, stream>>>(xs, label, flag, out_cq);
}
